// FP8Linear_51900384804990
// MI455X (gfx1250) — matrix operand order repaired and checked by reading, not compiled, not run on hardware
//
#include <hip/hip_runtime.h>

// ---------------------------------------------------------------------------
// FP8-style linear: out[M,N] = x[M,K] @ (w[N,K] * scale[N])^T
//                            = (x[M,K] @ w[N,K]^T) * scale[N]   (scale hoisted)
// M=8192, N=4096, K=4096 (divisible by all tile dims; no edge guards needed).
// bf16 WMMA (V_WMMA_F32_16X16X32_BF16), fp32 accumulate; scale applied in fp32
// at writeback (per output column), removing 32 VALU muls per K-iteration.
// BK=64: 16 WMMAs per barrier pair.
// ---------------------------------------------------------------------------

typedef __attribute__((ext_vector_type(16))) __bf16 v16bf;
typedef __attribute__((ext_vector_type(8)))  float  v8f;

union FragBF {
    uint4 q[2];   // 8 VGPRs = 16 bf16 per lane
    v16bf bf;
};

__device__ inline unsigned pack_bf16(float a, float b) {
#if __has_builtin(__builtin_amdgcn_cvt_pk_bf16_f32)
    typedef __attribute__((ext_vector_type(2))) __bf16 v2bf;
    v2bf r = __builtin_amdgcn_cvt_pk_bf16_f32(a, b);   // 1 VALU: v_cvt_pk_bf16_f32
    return __builtin_bit_cast(unsigned, r);
#else
    // round-to-nearest-even f32 -> bf16, packed pair {lo=a, hi=b}
    unsigned ua = __float_as_uint(a), ub = __float_as_uint(b);
    ua += 0x7FFFu + ((ua >> 16) & 1u);
    ub += 0x7FFFu + ((ub >> 16) & 1u);
    return (ua >> 16) | (ub & 0xFFFF0000u);
#endif
}

constexpr int BM = 128;         // M tile per block
constexpr int BN = 128;         // N tile per block
constexpr int BK = 64;          // K per staging pass (2 WMMA depths)
constexpr int LDS_STRIDE = 36;  // uints/row: 32 data pairs + 4 pad (144B, 16B-aligned)

__global__ __launch_bounds__(256)
void FP8Linear_wmma_bf16_kernel(const float* __restrict__ x,
                                const float* __restrict__ w,
                                const float* __restrict__ scale,
                                float* __restrict__ out,
                                int M, int N, int K)
{
    __shared__ __align__(16) unsigned int sA[BM * LDS_STRIDE]; // x tile, bf16 pairs
    __shared__ __align__(16) unsigned int sB[BN * LDS_STRIDE]; // w tile, bf16 pairs

    const int m0 = blockIdx.y * BM;
    const int n0 = blockIdx.x * BN;

    // --- staging: 256 threads = 128 rows x 2 halves (32 consecutive floats each)
    const int t     = threadIdx.x;
    const int srow  = t >> 1;
    const int shalf = t & 1;

    const float* gA = x + (size_t)(m0 + srow) * K + shalf * 32;
    const float* gB = w + (size_t)(n0 + srow) * K + shalf * 32;

    // --- compute: 8 waves in 2(M) x 4(N) grid; each wave: 64x32 = 4x2 WMMA tiles
    const int lane = threadIdx.x & 31;
    const int wave = threadIdx.x >> 5;
    const int wm   = wave >> 2;          // 0..1 -> 64-row half
    const int wn   = wave & 3;           // 0..3 -> 32-col strip
    const int half = lane >> 4;          // lane group (K-split per ISA layout)
    const int mrow = lane & 15;          // A row / B column within 16

    v8f acc[4][2];
#pragma unroll
    for (int i = 0; i < 4; ++i)
#pragma unroll
        for (int j = 0; j < 2; ++j)
            acc[i][j] = (v8f){0.f, 0.f, 0.f, 0.f, 0.f, 0.f, 0.f, 0.f};

    for (int k0 = 0; k0 < K; k0 += BK) {
        // ------------------- stage tiles: fp32 -> bf16 pairs in LDS ----------
        {
            const float4* pa = (const float4*)(gA + k0);
            const float4* pb = (const float4*)(gB + k0);

            float4 fa[8], fb[8];
#pragma unroll
            for (int j = 0; j < 8; ++j) { fa[j] = pa[j]; fb[j] = pb[j]; }

            // speculative prefetch of the next K tile (dropped if unmapped)
            __builtin_prefetch(gA + k0 + BK, 0, 3);
            __builtin_prefetch(gB + k0 + BK, 0, 3);

            const unsigned off = srow * LDS_STRIDE + shalf * 16;
#pragma unroll
            for (int j = 0; j < 4; ++j) {
                uint4 ua = {pack_bf16(fa[2 * j].x, fa[2 * j].y),
                            pack_bf16(fa[2 * j].z, fa[2 * j].w),
                            pack_bf16(fa[2 * j + 1].x, fa[2 * j + 1].y),
                            pack_bf16(fa[2 * j + 1].z, fa[2 * j + 1].w)};
                uint4 ub = {pack_bf16(fb[2 * j].x, fb[2 * j].y),
                            pack_bf16(fb[2 * j].z, fb[2 * j].w),
                            pack_bf16(fb[2 * j + 1].x, fb[2 * j + 1].y),
                            pack_bf16(fb[2 * j + 1].z, fb[2 * j + 1].w)};
                *(uint4*)&sA[off + 4 * j] = ua;
                *(uint4*)&sB[off + 4 * j] = ub;
            }
        }
        __syncthreads();

        // ------------------- fragments + WMMA: 2 K-depths per staging pass ---
#pragma unroll
        for (int ks = 0; ks < 2; ++ks) {
            // A layout (16x32 bf16): low lanes hold K pairs {0..3, 8..11},
            //                        high lanes hold K pairs {4..7, 12..15}.
            FragBF a[4];
#pragma unroll
            for (int mt = 0; mt < 4; ++mt) {
                const unsigned int* base =
                    sA + (wm * 64 + mt * 16 + mrow) * LDS_STRIDE + ks * 16 + half * 4;
                a[mt].q[0] = *(const uint4*)base;        // pairs half*4 .. +3
                a[mt].q[1] = *(const uint4*)(base + 8);  // pairs 8+half*4 .. +3
            }

            // B layout (32x16, B = w^T): low lanes K pairs {0..3, 8..11}, high lanes {4..7, 12..15};
            //                            column = lane%16.
            FragBF b[2];
#pragma unroll
            for (int nt = 0; nt < 2; ++nt) {
                const unsigned int* base =
                    sB + (wn * 32 + nt * 16 + mrow) * LDS_STRIDE + ks * 16 + half * 4;
                b[nt].q[0] = *(const uint4*)base;
                b[nt].q[1] = *(const uint4*)(base + 8);
            }

#pragma unroll
            for (int mt = 0; mt < 4; ++mt)
#pragma unroll
                for (int nt = 0; nt < 2; ++nt)
                    acc[mt][nt] = __builtin_amdgcn_wmma_f32_16x16x32_bf16(
                        /*neg_a=*/false, a[mt].bf,
                        /*neg_b=*/false, b[nt].bf,
                        /*c_mod=*/(short)0, acc[mt][nt],
                        /*reuse_a=*/false, /*reuse_b=*/false);
        }
        __syncthreads();
    }

    // ------------------- write back: apply per-column scale in fp32 -----------
    // C/D layout: VGPR r -> row r (lanes 0-15) / row r+8 (lanes 16-31), col = lane%16.
    const size_t sN = (size_t)N;
    const int col0 = n0 + wn * 32 + mrow;
    const float sc0 = scale[col0];        // nt = 0 column scale
    const float sc1 = scale[col0 + 16];   // nt = 1 column scale

#pragma unroll
    for (int mt = 0; mt < 4; ++mt) {
        // hoist the 64-bit row-base multiply out of the per-element loop
        float* prow = out + (size_t)(m0 + wm * 64 + mt * 16 + half * 8) * sN;
#pragma unroll
        for (int nt = 0; nt < 2; ++nt) {
            const int   col = col0 + nt * 16;
            const float sc  = nt ? sc1 : sc0;
#pragma unroll
            for (int r = 0; r < 8; ++r)
                prow[(size_t)r * sN + col] = acc[mt][nt][r] * sc;
        }
    }
}

extern "C" void kernel_launch(void* const* d_in, const int* in_sizes, int n_in,
                              void* d_out, int out_size, void* d_ws, size_t ws_size,
                              hipStream_t stream) {
    (void)n_in; (void)out_size; (void)d_ws; (void)ws_size;
    const float* x  = (const float*)d_in[0];
    const float* w  = (const float*)d_in[1];
    const float* sc = (const float*)d_in[2];
    float* out = (float*)d_out;

    const int       N = in_sizes[2];                       // D_OUT (scale has D_OUT elems)
    const long long wN = in_sizes[1];
    const int       K = (int)(wN / N);                     // D_IN
    const int       M = (int)((long long)in_sizes[0] / K); // B*S

    dim3 grid(N / BN, M / BM);   // N-fastest: x strips reused in L2
    FP8Linear_wmma_bf16_kernel<<<grid, 256, 0, stream>>>(x, w, sc, out, M, N, K);
}
